// NTXcent_66967130079810
// MI455X (gfx1250) — matrix operand order repaired and checked by reading, not compiled, not run on hardware
//
#include <hip/hip_runtime.h>
#include <math.h>

typedef __attribute__((ext_vector_type(16))) _Float16 v16h;
typedef __attribute__((ext_vector_type(8)))  _Float16 v8h;
typedef __attribute__((ext_vector_type(8)))  float    v8f;

#define NROWS 8192
#define DIM   256
#define INV_T 20.0f          // 1 / 0.05
#define MAXL  20.0f          // logits <= cos/T <= 20; fixed logsumexp shift
#define LOG2E 1.44269504088896340736f
#define EPSN  1e-8f

#define WAVES_PER_BLOCK 8
#define ROWS_PER_BLOCK  128   // 8 waves * 16 rows
#define COL_SEGS        8
#define COLS_PER_SEG    (NROWS / COL_SEGS)   // 1024
#define STAGE_COLS      32
#define NSTAGES         (COLS_PER_SEG / STAGE_COLS)  // 32
#define LDS_STRIDE      264   // 256 halves + 8 pad -> 528B stride, conflict-free b128 reads

static __device__ inline v16h combine16(v8h lo, v8h hi) {
    v16h r;
#pragma unroll
    for (int i = 0; i < 8; ++i) { r[i] = lo[i]; r[i + 8] = hi[i]; }
    return r;
}

static __device__ inline void wait_async0() {
#if __has_builtin(__builtin_amdgcn_s_wait_asynccnt)
    __builtin_amdgcn_s_wait_asynccnt(0);
#else
    asm volatile("s_wait_asynccnt 0x0" ::: "memory");
#endif
}

// ---------------- 1) row L2-normalize, f32 -> f16 ----------------
__global__ __launch_bounds__(256) void normalize_kernel(const float* __restrict__ x,
                                                        _Float16* __restrict__ xh) {
    int wave = threadIdx.x >> 5;
    int lane = threadIdx.x & 31;
    int row  = blockIdx.x * WAVES_PER_BLOCK + wave;
    const float* xr = x + (size_t)row * DIM;

    float4 a = ((const float4*)xr)[lane * 2 + 0];
    float4 b = ((const float4*)xr)[lane * 2 + 1];
    float s = a.x*a.x + a.y*a.y + a.z*a.z + a.w*a.w
            + b.x*b.x + b.y*b.y + b.z*b.z + b.w*b.w;
#pragma unroll
    for (int m = 16; m >= 1; m >>= 1) s += __shfl_xor(s, m, 32);
    float scale = 1.0f / fmaxf(sqrtf(s), EPSN);

    v8h h;
    h[0] = (_Float16)(a.x*scale); h[1] = (_Float16)(a.y*scale);
    h[2] = (_Float16)(a.z*scale); h[3] = (_Float16)(a.w*scale);
    h[4] = (_Float16)(b.x*scale); h[5] = (_Float16)(b.y*scale);
    h[6] = (_Float16)(b.z*scale); h[7] = (_Float16)(b.w*scale);
    ((v8h*)(xh + (size_t)row * DIM))[lane] = h;
}

// ---------------- 2) zero scratch accumulators ----------------
__global__ void zero_kernel(float* __restrict__ p, int n) {
    int i = blockIdx.x * blockDim.x + threadIdx.x;
    if (i < n) p[i] = 0.0f;
}

// ---------------- 3) fused  Xn@Xn^T  +  shifted sum-exp  ----------------
__global__ __launch_bounds__(256) void ntxent_main_kernel(const _Float16* __restrict__ xh,
                                                          float* __restrict__ sums,
                                                          float* __restrict__ pos) {
    __shared__ _Float16 lds[2][STAGE_COLS * LDS_STRIDE];   // double buffer, ~33 KB

    int lane    = threadIdx.x & 31;
    int wave    = threadIdx.x >> 5;
    int lm      = lane & 15;        // column slot / A-row slot
    int halfSel = lane >> 4;        // 0: lanes 0-15, 1: lanes 16-31
    int rowBase = blockIdx.x * ROWS_PER_BLOCK + wave * 16;
    int segBase = blockIdx.y * COLS_PER_SEG;

    // per-thread chunk for cooperative staging: 1024 chunks of 8 halves, 4 per thread
    // async copy memory -> LDS, tracked with ASYNCcnt (no VGPR staging)
    auto issue_stage = [&](int stage, int buf) {
        int cb = segBase + stage * STAGE_COLS;
#pragma unroll
        for (int i = 0; i < 4; ++i) {
            int lidx = (int)threadIdx.x + i * 256;
            int c    = lidx >> 5;
            int k8   = lidx & 31;
            unsigned long long gaddr =
                (unsigned long long)(size_t)(xh + (size_t)(cb + c) * DIM + k8 * 8);
            unsigned laddr = (unsigned)(size_t)(&lds[buf][c * LDS_STRIDE + k8 * 8]);
            asm volatile("global_load_async_to_lds_b128 %0, %1, off"
                         :: "v"(laddr), "v"(gaddr) : "memory");
        }
    };

    // ---- preload A fragments for this wave's 16 rows, full K=256 (8 chunks of 32) ----
    // ISA layout: lanes 0-15 hold K{0..7,16..23}, lanes 16-31 hold K{8..15,24..31}
    const _Float16* ap = xh + (size_t)(rowBase + lm) * DIM;
    v16h afrag[8];
#pragma unroll
    for (int kk = 0; kk < 8; ++kk) {
        v8h lo = *(const v8h*)(ap + kk * 32 +      halfSel * 8);
        v8h hi = *(const v8h*)(ap + kk * 32 + 16 + halfSel * 8);
        afrag[kk] = combine16(lo, hi);
    }

    float    ssum[8];
    float    pval[8];
    unsigned hit = 0u;
#pragma unroll
    for (int v = 0; v < 8; ++v) { ssum[v] = 0.0f; pval[v] = 0.0f; }

    issue_stage(0, 0);   // prologue

    for (int stage = 0; stage < NSTAGES; ++stage) {
        int buf = stage & 1;
        // own async loads for this stage have landed in LDS:
        wait_async0();
        // all waves' loads landed AND all waves done reading the other buffer:
        __syncthreads();
        // prefetch next stage into the other buffer, hidden under compute below
        if (stage + 1 < NSTAGES) issue_stage(stage + 1, buf ^ 1);

        int cb = segBase + stage * STAGE_COLS;
#pragma unroll
        for (int ct = 0; ct < 2; ++ct) {
            v8f acc = {};
            // B (=Xn^T): lane n holds column (ct*16+lm); lanes 0-15 K{0..7,16..23}, 16-31 K{8..15,24..31}
            const _Float16* bp = &lds[buf][(ct * 16 + lm) * LDS_STRIDE + halfSel * 8];
#pragma unroll
            for (int kk = 0; kk < 8; ++kk) {
                v8h lo = *(const v8h*)(bp + kk * 32);
                v8h hi = *(const v8h*)(bp + kk * 32 + 16);
                v16h bfrag = combine16(lo, hi);
                acc = __builtin_amdgcn_wmma_f32_16x16x32_f16(
                        false, afrag[kk], false, bfrag, (short)0, acc, false, false);
            }
            int col = cb + ct * 16 + lm;
            // C layout: VGPR v -> row = rowBase + 8*halfSel + v, N = lm
#pragma unroll
            for (int v = 0; v < 8; ++v) {
                int   r = rowBase + halfSel * 8 + v;
                float l = acc[v] * INV_T;
                if (col == (r ^ 1)) { pval[v] = l; hit |= (1u << v); }
                float e = (col == r) ? 0.0f
                                     : __builtin_amdgcn_exp2f((l - MAXL) * LOG2E);
                ssum[v] += e;
            }
        }
    }

    // reduce partial sums across the 16 column-slot lanes of each half-wave
#pragma unroll
    for (int v = 0; v < 8; ++v) {
        float s = ssum[v];
        s += __shfl_xor(s, 1, 32);
        s += __shfl_xor(s, 2, 32);
        s += __shfl_xor(s, 4, 32);
        s += __shfl_xor(s, 8, 32);
        int r = rowBase + halfSel * 8 + v;
        if (lm == 0) atomicAdd(&sums[r], s);
        if (hit & (1u << v)) pos[r] = pval[v];   // exactly one writer per row, grid-wide
    }
}

// ---------------- 4) finalize: mean(lse - pos) ----------------
__global__ __launch_bounds__(256) void finalize_kernel(const float* __restrict__ sums,
                                                       const float* __restrict__ pos,
                                                       float* __restrict__ out) {
    __shared__ float red[256];
    float acc = 0.0f;
    for (int i = threadIdx.x; i < NROWS; i += 256)
        acc += (MAXL + logf(sums[i])) - pos[i];
    red[threadIdx.x] = acc;
    __syncthreads();
    for (int s = 128; s > 0; s >>= 1) {
        if (threadIdx.x < (unsigned)s) red[threadIdx.x] += red[threadIdx.x + s];
        __syncthreads();
    }
    if (threadIdx.x == 0) out[0] = red[0] / (float)NROWS;
}

extern "C" void kernel_launch(void* const* d_in, const int* in_sizes, int n_in,
                              void* d_out, int out_size, void* d_ws, size_t ws_size,
                              hipStream_t stream) {
    (void)in_sizes; (void)n_in; (void)out_size; (void)ws_size;
    const float* x = (const float*)d_in[0];
    float* out = (float*)d_out;

    _Float16* xh   = (_Float16*)d_ws;                                   // 4 MB
    float*    sums = (float*)((char*)d_ws + (size_t)NROWS * DIM * 2);   // 32 KB
    float*    pos  = sums + NROWS;                                      // 32 KB

    normalize_kernel<<<NROWS / WAVES_PER_BLOCK, 256, 0, stream>>>(x, xh);
    zero_kernel<<<(2 * NROWS + 255) / 256, 256, 0, stream>>>(sums, 2 * NROWS);
    dim3 grid(NROWS / ROWS_PER_BLOCK, COL_SEGS);
    ntxent_main_kernel<<<grid, 256, 0, stream>>>(xh, sums, pos);
    finalize_kernel<<<1, 256, 0, stream>>>(sums, pos, out);
}
